// MultiHeadAttentionWithRelativePosition_52115133169812
// MI455X (gfx1250) — matrix operand order repaired and checked by reading, not compiled, not run on hardware
//
#include <hip/hip_runtime.h>
#include <hip/hip_bf16.h>
#include <math.h>

typedef __bf16 bf16;
typedef __attribute__((ext_vector_type(16))) __bf16 bf16x16;
typedef __attribute__((ext_vector_type(8)))  __bf16 bf16x8;
typedef __attribute__((ext_vector_type(8)))  float  f32x8;
typedef __attribute__((ext_vector_type(4)))  float  f32x4;

#define D_MODEL   1024
#define NUM_HEADS 16
#define D_K       64
#define SEQ       768
#define BATCH     8
#define ROWS      (BATCH*SEQ)      /* 6144 */
#define NREL      1535             /* 2*768-1 */

__device__ __forceinline__ bf16 f2bf(float f) { return (bf16)f; }

__device__ __forceinline__ f32x8 zero8() {
  f32x8 z; for (int e = 0; e < 8; ++e) z[e] = 0.0f; return z;
}

__device__ __forceinline__ f32x8 wmma_bf16(bf16x16 a, bf16x16 b, f32x8 c) {
  return __builtin_amdgcn_wmma_f32_16x16x32_bf16(false, a, false, b, (short)0, c,
                                                 false, false);
}

__device__ __forceinline__ bf16x16 cat16(bf16x8 lo, bf16x8 hi) {
  return __builtin_shufflevector(lo, hi, 0,1,2,3,4,5,6,7,8,9,10,11,12,13,14,15);
}

// CDNA5 async global->LDS copy (ASYNCcnt-tracked, §15.18 op 98)
__device__ __forceinline__ void async_copy_b128(void* ldsDst, const void* gSrc) {
  unsigned lds_off = (unsigned)(uintptr_t)ldsDst;   // AS3 offset = low 32 bits
  asm volatile("global_load_async_to_lds_b128 %0, %1, off"
               :: "v"(lds_off), "v"(gSrc) : "memory");
}
__device__ __forceinline__ void wait_async() {
  asm volatile("s_wait_asynccnt 0x0" ::: "memory");
}

// ---------------------------------------------------------------------------
// Kernel 1: projection GEMM  out = (X @ W^T + b) * scaleOut  (bf16, head-major)
// 128(M) x 64(N) tile / 256-thread WG, K-step 32, double-buffered LDS.
// ---------------------------------------------------------------------------
__global__ __launch_bounds__(256) void proj_gemm_kernel(
    const float* __restrict__ X, const float* __restrict__ W,
    const float* __restrict__ bias, bf16* __restrict__ out, float scaleOut)
{
  __shared__ bf16 As[2][128][40];   // [buf][m][k]
  __shared__ bf16 Bs[2][64][40];    // [buf][n][k]

  const int tid  = threadIdx.x;
  const int wave = tid >> 5;
  const int lane = tid & 31;
  const int l16  = lane & 15;
  const int hf   = lane >> 4;
  const int row0 = blockIdx.y * 128;
  const int n0   = blockIdx.x * 64;

  const int ar_ = tid >> 1;             // A staging row
  const int akh = (tid & 1) * 16;       // A staging k-offset
  const int bn_ = tid >> 2;             // B staging n-row
  const int bkq = (tid & 3) * 8;        // B staging k-offset

  f32x8 acc[4];
  for (int i = 0; i < 4; ++i) acc[i] = zero8();

  f32x4 arv[4];   // staged A values (16 f32)
  f32x4 brv[2];   // staged B values (8 f32)

  // issue+commit tile 0
  {
    const float* src = X + (size_t)(row0 + ar_) * D_MODEL + akh;
    for (int j = 0; j < 4; ++j) arv[j] = *(const f32x4*)(src + 4 * j);
    const float* ws = W + (size_t)(n0 + bn_) * D_MODEL + bkq;
    brv[0] = *(const f32x4*)(ws + 0);
    brv[1] = *(const f32x4*)(ws + 4);
    bf16* ad = &As[0][ar_][akh];
    for (int j = 0; j < 4; ++j)
      for (int e = 0; e < 4; ++e) ad[4 * j + e] = f2bf(arv[j][e]);
    bf16* bd = &Bs[0][bn_][bkq];
    for (int e = 0; e < 4; ++e) { bd[e] = f2bf(brv[0][e]); bd[4 + e] = f2bf(brv[1][e]); }
  }

  int buf = 0;
  for (int k0 = 0; k0 < D_MODEL; k0 += 32) {
    __syncthreads();
    const bool more = (k0 + 32) < D_MODEL;
    if (more) {   // issue next tile's global loads (latency hidden under WMMA)
      const float* src = X + (size_t)(row0 + ar_) * D_MODEL + (k0 + 32) + akh;
      __builtin_prefetch(src + 32, 0, 0);
      for (int j = 0; j < 4; ++j) arv[j] = *(const f32x4*)(src + 4 * j);
      const float* ws = W + (size_t)(n0 + bn_) * D_MODEL + (k0 + 32) + bkq;
      brv[0] = *(const f32x4*)(ws + 0);
      brv[1] = *(const f32x4*)(ws + 4);
    }

    // compute on current buffer: batch all fragment loads, then 4 WMMAs
    bf16x8 alo = *(const bf16x8*)&As[buf][wave * 16 + l16][hf * 16];
    bf16x8 ahi = *(const bf16x8*)&As[buf][wave * 16 + l16][hf * 16 + 8];
    bf16x8 blo[4], bhi[4];
#pragma unroll
    for (int ni = 0; ni < 4; ++ni) {
      blo[ni] = *(const bf16x8*)&Bs[buf][ni * 16 + l16][hf * 16];
      bhi[ni] = *(const bf16x8*)&Bs[buf][ni * 16 + l16][hf * 16 + 8];
    }
    bf16x16 afrag = cat16(alo, ahi);
#pragma unroll
    for (int ni = 0; ni < 4; ++ni)
      acc[ni] = wmma_bf16(afrag, cat16(blo[ni], bhi[ni]), acc[ni]);

    if (more) {   // commit next tile into alternate buffer
      bf16* ad = &As[buf ^ 1][ar_][akh];
      for (int j = 0; j < 4; ++j)
        for (int e = 0; e < 4; ++e) ad[4 * j + e] = f2bf(arv[j][e]);
      bf16* bd = &Bs[buf ^ 1][bn_][bkq];
      for (int e = 0; e < 4; ++e) { bd[e] = f2bf(brv[0][e]); bd[4 + e] = f2bf(brv[1][e]); }
    }
    buf ^= 1;
  }

  // epilogue: head-major bf16 store.  128-row tile lies in one batch element.
  const int bb = row0 / SEQ;                 // block-uniform (SALU)
  const int lb = row0 - bb * SEQ;
#pragma unroll
  for (int ni = 0; ni < 4; ++ni) {
    const int n  = n0 + ni * 16 + l16;
    const int hh = n >> 6, d = n & 63;
    const float bsv = bias[n];
    bf16* obase = out + (((size_t)(bb * NUM_HEADS + hh)) * SEQ) * D_K + d;
#pragma unroll
    for (int v = 0; v < 8; ++v) {
      int l = lb + wave * 16 + v + hf * 8;
      obase[(size_t)l * D_K] = f2bf((acc[ni][v] + bsv) * scaleOut);
    }
  }
}

// ---------------------------------------------------------------------------
// Kernel 2: flash attention with relative position bias.
// One WG (128 thr, 4 waves) per (b, h, 64-row q-block). Each wave owns 16
// complete query rows -> softmax is wave-local. Rel bias via QE = Qblk @ E^T
// (WMMA) into LDS. Q/K tiles staged with global_load_async_to_lds_b128.
// ---------------------------------------------------------------------------
__global__ __launch_bounds__(128) void attn_kernel(
    const bf16* __restrict__ qp, const bf16* __restrict__ kp,
    const bf16* __restrict__ vp, const float* __restrict__ Erel,
    bf16* __restrict__ attn_out)
{
  __shared__ bf16 Qs[64][72];        // [qrow][dk]
  __shared__ bf16 Kt[64][72];        // [key][dk]
  __shared__ bf16 Vt[64][72];        // [dk][key]  (transposed)
  __shared__ bf16 Ps[4][16][72];     // per-wave P relayout buffer
  __shared__ bf16 QE[64][840];       // [qrow][ecol], 832 cols used

  const int tid  = threadIdx.x;
  const int wave = tid >> 5;     // 0..3
  const int lane = tid & 31;
  const int l16  = lane & 15;
  const int hf   = lane >> 4;
  const int q0   = blockIdx.x * 64;
  const int h    = blockIdx.y;
  const int b    = blockIdx.z;
  const int e0b  = 704 - q0;     // first E row needed by this q-block
  const int mr   = wave * 16;    // this wave's query-row base in the block

  const bf16* qg = qp + (((size_t)(b * NUM_HEADS + h)) * SEQ + q0) * D_K;
  const bf16* kg = kp + ((size_t)(b * NUM_HEADS + h)) * SEQ * D_K;
  const bf16* vg = vp + ((size_t)(b * NUM_HEADS + h)) * SEQ * D_K;

  const int sr = tid >> 1;            // staging row 0..63
  const int sd = (tid & 1) * 32;      // staging dk offset

  { // stage Q block asynchronously (straight bf16 copy)
    const bf16* src = qg + (size_t)sr * D_K + sd;
    for (int j = 0; j < 4; ++j)
      async_copy_b128(&Qs[sr][sd + j * 8], src + j * 8);
    wait_async();
  }
  __syncthreads();

  // preload this wave's Q A-fragments (dk = 2 K-steps of 32)
  bf16x16 aq[2];
#pragma unroll
  for (int ks = 0; ks < 2; ++ks) {
    bf16x8 lo = *(const bf16x8*)&Qs[mr + l16][ks * 32 + hf * 16];
    bf16x8 hi = *(const bf16x8*)&Qs[mr + l16][ks * 32 + hf * 16 + 8];
    aq[ks] = cat16(lo, hi);
  }

  // Phase 1: QE = Qblk @ E_window^T  (52 column subtiles of 16)
  for (int c = 0; c < 52; ++c) {
    f32x8 accE = zero8();
#pragma unroll
    for (int ks = 0; ks < 2; ++ks) {
      int ecol = e0b + c * 16 + l16;
      if (ecol > NREL - 1) ecol = NREL - 1;
      const float* er = Erel + (size_t)ecol * D_K + ks * 32 + hf * 16;
      f32x4 e0v = *(const f32x4*)(er + 0);
      f32x4 e1v = *(const f32x4*)(er + 4);
      f32x4 e2v = *(const f32x4*)(er + 8);
      f32x4 e3v = *(const f32x4*)(er + 12);
      bf16x16 bfr;
      for (int e = 0; e < 4; ++e) {
        bfr[e]      = f2bf(e0v[e]); bfr[4  + e] = f2bf(e1v[e]);
        bfr[8 + e]  = f2bf(e2v[e]); bfr[12 + e] = f2bf(e3v[e]);
      }
      accE = wmma_bf16(aq[ks], bfr, accE);
    }
#pragma unroll
    for (int v = 0; v < 8; ++v)
      QE[mr + v + hf * 8][c * 16 + l16] = f2bf(accE[v]);
  }
  __syncthreads();

  // flash state (per lane: 8 rows = v + 8*hf within this wave's 16 rows)
  float mrow[8], lrow[8];
  for (int v = 0; v < 8; ++v) { mrow[v] = -1e30f; lrow[v] = 0.0f; }
  f32x8 accO[4];
  for (int i = 0; i < 4; ++i) accO[i] = zero8();

  for (int kt = 0; kt < 12; ++kt) {
    const int k0 = kt * 64;
    __syncthreads();
    { // K tile: async copy; V tile: register transpose into [dk][key]
      const bf16* ks_ = kg + (size_t)(k0 + sr) * D_K + sd;
      for (int j = 0; j < 4; ++j)
        async_copy_b128(&Kt[sr][sd + j * 8], ks_ + j * 8);
      const bf16* vs_ = vg + (size_t)(k0 + sr) * D_K + sd;
      bf16x8 vv[4];
      for (int j = 0; j < 4; ++j) vv[j] = *(const bf16x8*)(vs_ + j * 8);
      for (int j = 0; j < 4; ++j)
        for (int e = 0; e < 8; ++e) Vt[sd + j * 8 + e][sr] = vv[j][e];
      wait_async();
    }
    __syncthreads();

    // S = Q @ K^T (content, pre-scaled via q): batch loads, then WMMAs
    bf16x8 klo[4][2], khi[4][2];
#pragma unroll
    for (int ni = 0; ni < 4; ++ni)
#pragma unroll
      for (int ks = 0; ks < 2; ++ks) {
        klo[ni][ks] = *(const bf16x8*)&Kt[ni * 16 + l16][ks * 32 + hf * 16];
        khi[ni][ks] = *(const bf16x8*)&Kt[ni * 16 + l16][ks * 32 + hf * 16 + 8];
      }
    f32x8 sacc[4];
#pragma unroll
    for (int ni = 0; ni < 4; ++ni) {
      sacc[ni] = zero8();
#pragma unroll
      for (int ks = 0; ks < 2; ++ks)
        sacc[ni] = wmma_bf16(aq[ks], cat16(klo[ni][ks], khi[ni][ks]), sacc[ni]);
    }

    // + relative-position bias gathered from QE (clip is a no-op for L=768)
#pragma unroll
    for (int ni = 0; ni < 4; ++ni)
#pragma unroll
      for (int v = 0; v < 8; ++v) {
        int ecol = k0 + ni * 16 + l16 - mr - v - hf * 8 + 63;
        sacc[ni][v] += (float)QE[mr + v + hf * 8][ecol];
      }

    // wave-local flash softmax update
    float alpha[8];
#pragma unroll
    for (int v = 0; v < 8; ++v) {
      float t = sacc[0][v];
      for (int ni = 1; ni < 4; ++ni) t = fmaxf(t, sacc[ni][v]);
      for (int off = 8; off >= 1; off >>= 1)
        t = fmaxf(t, __shfl_xor(t, off, 32));
      float mnew = fmaxf(mrow[v], t);
      alpha[v] = __expf(mrow[v] - mnew);
      mrow[v] = mnew;
    }
#pragma unroll
    for (int v = 0; v < 8; ++v) {
      float s = 0.0f;
      for (int ni = 0; ni < 4; ++ni) {
        float p = __expf(sacc[ni][v] - mrow[v]);
        sacc[ni][v] = p; s += p;
      }
      for (int off = 8; off >= 1; off >>= 1)
        s += __shfl_xor(s, off, 32);
      lrow[v] = lrow[v] * alpha[v] + s;
    }
#pragma unroll
    for (int ni = 0; ni < 4; ++ni)
      for (int v = 0; v < 8; ++v) accO[ni][v] *= alpha[v];

    // relayout P (C-layout) -> LDS -> A-fragments (wave-private, DS in-order)
#pragma unroll
    for (int ni = 0; ni < 4; ++ni)
      for (int v = 0; v < 8; ++v)
        Ps[wave][v + hf * 8][ni * 16 + l16] = f2bf(sacc[ni][v]);

    // O += P @ V
#pragma unroll
    for (int ks = 0; ks < 2; ++ks) {
      bf16x8 plo = *(const bf16x8*)&Ps[wave][l16][ks * 32 + hf * 16];
      bf16x8 phi = *(const bf16x8*)&Ps[wave][l16][ks * 32 + hf * 16 + 8];
      bf16x8 vlo[4], vhi[4];
#pragma unroll
      for (int nio = 0; nio < 4; ++nio) {
        vlo[nio] = *(const bf16x8*)&Vt[nio * 16 + l16][ks * 32 + hf * 16];
        vhi[nio] = *(const bf16x8*)&Vt[nio * 16 + l16][ks * 32 + hf * 16 + 8];
      }
      bf16x16 ap = cat16(plo, phi);
#pragma unroll
      for (int nio = 0; nio < 4; ++nio)
        accO[nio] = wmma_bf16(ap, cat16(vlo[nio], vhi[nio]), accO[nio]);
    }
  }

  // epilogue: normalize, store merged-head bf16 [row = b*768+q][col = h*64+dk]
#pragma unroll
  for (int nio = 0; nio < 4; ++nio)
#pragma unroll
    for (int v = 0; v < 8; ++v) {
      int r  = b * SEQ + q0 + mr + v + hf * 8;
      int cc = h * D_K + nio * 16 + l16;
      float o = accO[nio][v] / lrow[v];
      attn_out[(size_t)r * D_MODEL + cc] = f2bf(o);
    }
}

// ---------------------------------------------------------------------------
// Kernel 3: output projection  out = A(bf16) @ W_o^T + b_o  -> fp32 d_out
// Double-buffered like kernel 1; A is already bf16.
// ---------------------------------------------------------------------------
__global__ __launch_bounds__(256) void oproj_gemm_kernel(
    const bf16* __restrict__ A, const float* __restrict__ W,
    const float* __restrict__ bias, float* __restrict__ out)
{
  __shared__ bf16 As[2][128][40];
  __shared__ bf16 Bs[2][64][40];

  const int tid  = threadIdx.x;
  const int wave = tid >> 5;
  const int lane = tid & 31;
  const int l16  = lane & 15;
  const int hf   = lane >> 4;
  const int row0 = blockIdx.y * 128;
  const int n0   = blockIdx.x * 64;

  const int ar_ = tid >> 1;
  const int akh = (tid & 1) * 16;
  const int bn_ = tid >> 2;
  const int bkq = (tid & 3) * 8;

  f32x8 acc[4];
  for (int i = 0; i < 4; ++i) acc[i] = zero8();

  bf16x8 arv[2];
  f32x4  brv[2];

  { // issue+commit tile 0
    const bf16* src = A + (size_t)(row0 + ar_) * D_MODEL + akh;
    arv[0] = *(const bf16x8*)(src);
    arv[1] = *(const bf16x8*)(src + 8);
    const float* ws = W + (size_t)(n0 + bn_) * D_MODEL + bkq;
    brv[0] = *(const f32x4*)(ws + 0);
    brv[1] = *(const f32x4*)(ws + 4);
    *(bf16x8*)&As[0][ar_][akh]     = arv[0];
    *(bf16x8*)&As[0][ar_][akh + 8] = arv[1];
    bf16* bd = &Bs[0][bn_][bkq];
    for (int e = 0; e < 4; ++e) { bd[e] = f2bf(brv[0][e]); bd[4 + e] = f2bf(brv[1][e]); }
  }

  int buf = 0;
  for (int k0 = 0; k0 < D_MODEL; k0 += 32) {
    __syncthreads();
    const bool more = (k0 + 32) < D_MODEL;
    if (more) {
      const bf16* src = A + (size_t)(row0 + ar_) * D_MODEL + (k0 + 32) + akh;
      arv[0] = *(const bf16x8*)(src);
      arv[1] = *(const bf16x8*)(src + 8);
      const float* ws = W + (size_t)(n0 + bn_) * D_MODEL + (k0 + 32) + bkq;
      brv[0] = *(const f32x4*)(ws + 0);
      brv[1] = *(const f32x4*)(ws + 4);
    }

    bf16x8 alo = *(const bf16x8*)&As[buf][wave * 16 + l16][hf * 16];
    bf16x8 ahi = *(const bf16x8*)&As[buf][wave * 16 + l16][hf * 16 + 8];
    bf16x8 blo[4], bhi[4];
#pragma unroll
    for (int ni = 0; ni < 4; ++ni) {
      blo[ni] = *(const bf16x8*)&Bs[buf][ni * 16 + l16][hf * 16];
      bhi[ni] = *(const bf16x8*)&Bs[buf][ni * 16 + l16][hf * 16 + 8];
    }
    bf16x16 afrag = cat16(alo, ahi);
#pragma unroll
    for (int ni = 0; ni < 4; ++ni)
      acc[ni] = wmma_bf16(afrag, cat16(blo[ni], bhi[ni]), acc[ni]);

    if (more) {
      *(bf16x8*)&As[buf ^ 1][ar_][akh]     = arv[0];
      *(bf16x8*)&As[buf ^ 1][ar_][akh + 8] = arv[1];
      bf16* bd = &Bs[buf ^ 1][bn_][bkq];
      for (int e = 0; e < 4; ++e) { bd[e] = f2bf(brv[0][e]); bd[4 + e] = f2bf(brv[1][e]); }
    }
    buf ^= 1;
  }

#pragma unroll
  for (int ni = 0; ni < 4; ++ni) {
    const int n = n0 + ni * 16 + l16;
    const float bsv = bias[n];
#pragma unroll
    for (int v = 0; v < 8; ++v) {
      int r = row0 + wave * 16 + v + hf * 8;
      out[(size_t)r * D_MODEL + n] = acc[ni][v] + bsv;
    }
  }
}

// ---------------------------------------------------------------------------
extern "C" void kernel_launch(void* const* d_in, const int* in_sizes, int n_in,
                              void* d_out, int out_size, void* d_ws, size_t ws_size,
                              hipStream_t stream) {
  (void)in_sizes; (void)n_in; (void)out_size; (void)ws_size;
  const float* query = (const float*)d_in[0];
  const float* key   = (const float*)d_in[1];
  const float* value = (const float*)d_in[2];
  const float* w_q   = (const float*)d_in[3];
  const float* b_q   = (const float*)d_in[4];
  const float* w_k   = (const float*)d_in[5];
  const float* b_k   = (const float*)d_in[6];
  const float* w_v   = (const float*)d_in[7];
  const float* b_v   = (const float*)d_in[8];
  const float* w_o   = (const float*)d_in[9];
  const float* b_o   = (const float*)d_in[10];
  const float* erel  = (const float*)d_in[11];

  const size_t PROJ_ELEMS = (size_t)ROWS * D_MODEL;   // 6,291,456 bf16 each
  bf16* qp = (bf16*)d_ws;
  bf16* kp = qp + PROJ_ELEMS;
  bf16* vp = kp + PROJ_ELEMS;
  bf16* ao = vp + PROJ_ELEMS;
  float* out = (float*)d_out;

  dim3 gridP(D_MODEL / 64, ROWS / 128), blockP(256);
  const float scale = 0.125f;  // 1/sqrt(64), folded into Q projection
  proj_gemm_kernel<<<gridP, blockP, 0, stream>>>(query, w_q, b_q, qp, scale);
  proj_gemm_kernel<<<gridP, blockP, 0, stream>>>(key,   w_k, b_k, kp, 1.0f);
  proj_gemm_kernel<<<gridP, blockP, 0, stream>>>(value, w_v, b_v, vp, 1.0f);

  dim3 gridA(SEQ / 64, NUM_HEADS, BATCH), blockA(128);
  attn_kernel<<<gridA, blockA, 0, stream>>>(qp, kp, vp, erel, ao);

  oproj_gemm_kernel<<<gridP, blockP, 0, stream>>>(ao, w_o, b_o, out);
}
